// StackedSPN_24730421690431
// MI455X (gfx1250) — matrix operand order repaired and checked by reading, not compiled, not run on hardware
//
#include <hip/hip_runtime.h>
#include <cstdint>
#include <cstddef>

typedef _Float16 half_t;
typedef __attribute__((ext_vector_type(16))) _Float16 v16h;
typedef __attribute__((ext_vector_type(8)))  float    v8f;
typedef __attribute__((ext_vector_type(4)))  float    f32x4;

#define NSAMP 512
// channels C = 16, cross dim = 256, per-layer features: 256,128,64,32

// ---------------------------------------------------------------------------
// Kernel 1: P[d][c][k] = softmax_k(W[d][k][c]) as f16, transposed so each
// WMMA B-row (fixed N=c) is K-contiguous.  W layout: [D][256][16] f32.
// One thread per (d,c) column.
// ---------------------------------------------------------------------------
__global__ __launch_bounds__(256) void softmax_w_kernel(
    const float* __restrict__ W, half_t* __restrict__ P, int D) {
  int tid = blockIdx.x * 256 + threadIdx.x;
  int d = tid >> 4;
  int c = tid & 15;
  if (d >= D) return;
  const float* wdc = W + (size_t)d * 256 * 16 + c;
  float m = -3.402823466e+38f;
  for (int k = 0; k < 256; ++k) m = fmaxf(m, wdc[k * 16]);
  float s = 0.f;
  for (int k = 0; k < 256; ++k) s += __expf(wdc[k * 16] - m);
  float logZ = m + __logf(s);
  half_t* pr = P + ((size_t)d * 16 + c) * 256;
  for (int k = 0; k < 256; ++k) pr[k] = (half_t)__expf(wdc[k * 16] - logZ);
}

// ---------------------------------------------------------------------------
// Kernel 2: one SPN layer (cross_product + sum_layer) as batched WMMA GEMM.
//   hin : [N][2D][16] f32, P : [D][16][256] f16, hout : [N][D][16] f32
// Block = 256 threads = 8 waves; wave w handles feature d = by*8+w for the
// 16-sample tile n0 = bx*16.
// Per wave:
//   phase A: lane L loads channel L (left c=L<16 / right c=L-16) of rows
//            m=0..15; half-wave shfl_xor max-reduce -> ml/mr; stage
//            e[m][chan]=exp(v - max) to LDS; mrow[m]=ml+mr.
//   phase B: per-lane register preload of e-row M=L&15 (eL[16], eR[8]);
//            8 x v_wmma_f32_16x16x32_f16 with A built in-register:
//            a[e] = f16( eL[2s + (e>>3)] * eR[e&7] )   (ISA A-layout)
//            b    = 16 B at k=32s+8h and 16 B at k=32s+16+8h of P row N=L&15 (ISA B-layout)
//   out[n0+M][d][c] = mrow[M] + log(acc)
// ---------------------------------------------------------------------------
__global__ __launch_bounds__(256) void spn_layer_kernel(
    const float* __restrict__ hin, const half_t* __restrict__ P,
    float* __restrict__ hout, int D) {
  constexpr int ES = 36;  // row stride (floats): 16-aligned, bank-spreading
  __shared__ __align__(16) float eArr[8][16][ES];
  __shared__ float mrow[8][16];

  const int wave = threadIdx.x >> 5;
  const int L    = threadIdx.x & 31;
  const int half = L >> 4;
  const int lo   = L & 15;
  const int d    = blockIdx.y * 8 + wave;
  const int n0   = blockIdx.x * 16;
  const int F    = 2 * D;

  // ---- phase A: exps + maxes into LDS ----
  const float* rowBase = hin + ((size_t)n0 * F + 2 * (size_t)d) * 16 + L;
  #pragma unroll
  for (int m = 0; m < 16; ++m) {
    float v = rowBase[(size_t)m * F * 16];
    float mx = v;
    #pragma unroll
    for (int off = 8; off >= 1; off >>= 1)
      mx = fmaxf(mx, __shfl_xor(mx, off, 32));      // max within 16-lane half
    float mo = __shfl_xor(mx, 16, 32);              // other half's max
    eArr[wave][m][L] = __expf(v - mx);
    if (L == 0) mrow[wave][m] = mx + mo;            // m[n] = max l + max r
  }

  // ---- per-lane register preload of its A row (M = lo) ----
  float eL[16];
  {
    const f32x4* rp = (const f32x4*)(&eArr[wave][lo][0]);
    #pragma unroll
    for (int t = 0; t < 4; ++t) {
      f32x4 q = rp[t];
      eL[4*t+0] = q[0]; eL[4*t+1] = q[1]; eL[4*t+2] = q[2]; eL[4*t+3] = q[3];
    }
  }
  float eR[8];
  {
    const f32x4* rp = (const f32x4*)(&eArr[wave][lo][16 + 8 * half]);
    f32x4 q0 = rp[0], q1 = rp[1];
    eR[0]=q0[0]; eR[1]=q0[1]; eR[2]=q0[2]; eR[3]=q0[3];
    eR[4]=q1[0]; eR[5]=q1[1]; eR[6]=q1[2]; eR[7]=q1[3];
  }

  // ---- phase B: 8 WMMAs over K = 256 ----
  const half_t* Pd = P + ((size_t)d * 16 + lo) * 256 + 8 * half;
  v8f acc = {};
  #pragma unroll
  for (int s = 0; s < 8; ++s) {
    f32x4 b0 = *(const f32x4*)(Pd + 32 * s), b1 = *(const f32x4*)(Pd + 32 * s + 16);   // 2x global_load_b128
    v16h b = __builtin_bit_cast(v16h, (v8f)__builtin_shufflevector(b0, b1, 0, 1, 2, 3, 4, 5, 6, 7));
    v16h a;
    #pragma unroll
    for (int e = 0; e < 16; ++e) {
      // A-layout: K = 32s + (e&7) + 8*half + 16*(e>>3)  =>  i=2s+(e>>3), j=8*half+(e&7)
      a[e] = (half_t)(eL[2 * s + (e >> 3)] * eR[e & 7]);
    }
    acc = __builtin_amdgcn_wmma_f32_16x16x32_f16(
        false, a, false, b, (short)0, acc, false, false);
  }

  // ---- epilogue: back to log domain ----
  float* outBase = hout + ((size_t)n0 * D + d) * 16 + lo;
  #pragma unroll
  for (int r = 0; r < 8; ++r) {
    int M = r + 8 * half;                   // C/D layout: lane half selects M+8
    outBase[(size_t)M * D * 16] = mrow[wave][M] + __logf(acc[r]);
  }
}

// ---------------------------------------------------------------------------
// Kernel 3: hsum[n][c] = sum_d h4[n][d][c] (D=32); out[n] = lse_c(hsum + lw),
// lw = log_softmax(W_root).  One wave per sample n; 8 waves per block.
// ---------------------------------------------------------------------------
__global__ __launch_bounds__(256) void root_kernel(
    const float* __restrict__ h4, const float* __restrict__ Wr,
    float* __restrict__ out) {
  const int wave = threadIdx.x >> 5;
  const int L    = threadIdx.x & 31;
  const int half = L >> 4;
  const int c    = L & 15;
  const int n    = blockIdx.x * 8 + wave;

  // log_softmax of the 16 root weights (both halves compute identically)
  float wv = Wr[c];
  float wm = wv;
  #pragma unroll
  for (int off = 8; off >= 1; off >>= 1) wm = fmaxf(wm, __shfl_xor(wm, off, 32));
  float wsum = __expf(wv - wm);
  #pragma unroll
  for (int off = 8; off >= 1; off >>= 1) wsum += __shfl_xor(wsum, off, 32);
  float lw = wv - (wm + __logf(wsum));

  // sum over 32 features, split across the two half-waves
  float p = 0.f;
  const float* base = h4 + (size_t)n * 512 + c;
  #pragma unroll
  for (int t = 0; t < 16; ++t) p += base[(half + 2 * t) * 16];
  p += __shfl_xor(p, 16, 32);

  float v = p + lw;
  float m2 = v;
  #pragma unroll
  for (int off = 8; off >= 1; off >>= 1) m2 = fmaxf(m2, __shfl_xor(m2, off, 32));
  float s2 = __expf(v - m2);
  #pragma unroll
  for (int off = 8; off >= 1; off >>= 1) s2 += __shfl_xor(s2, off, 32);
  if (L == 0) out[n] = m2 + __logf(s2);
}

// ---------------------------------------------------------------------------
extern "C" void kernel_launch(void* const* d_in, const int* in_sizes, int n_in,
                              void* d_out, int out_size, void* d_ws, size_t ws_size,
                              hipStream_t stream) {
  (void)in_sizes; (void)n_in; (void)out_size; (void)ws_size;
  const float* x  = (const float*)d_in[0];
  const float* W1 = (const float*)d_in[1];
  const float* W2 = (const float*)d_in[2];
  const float* W3 = (const float*)d_in[3];
  const float* W4 = (const float*)d_in[4];
  const float* Wr = (const float*)d_in[5];
  float* out = (float*)d_out;

  const int D1 = 256, D2 = 128, D3 = 64, D4 = 32;

  char* ws = (char*)d_ws;
  size_t off = 0;
  auto take = [&](size_t bytes) -> void* {
    void* p = ws + off;
    off += (bytes + 255) & ~(size_t)255;
    return p;
  };
  half_t* P1 = (half_t*)take((size_t)D1 * 16 * 256 * sizeof(half_t));
  half_t* P2 = (half_t*)take((size_t)D2 * 16 * 256 * sizeof(half_t));
  half_t* P3 = (half_t*)take((size_t)D3 * 16 * 256 * sizeof(half_t));
  half_t* P4 = (half_t*)take((size_t)D4 * 16 * 256 * sizeof(half_t));
  float*  h1 = (float*)take((size_t)NSAMP * D1 * 16 * sizeof(float));
  float*  h2 = (float*)take((size_t)NSAMP * D2 * 16 * sizeof(float));
  float*  h3 = (float*)take((size_t)NSAMP * D3 * 16 * sizeof(float));
  float*  h4 = (float*)take((size_t)NSAMP * D4 * 16 * sizeof(float));

  // weight softmax prologue (L2-resident f16 B matrices, ~3.8 MB total)
  softmax_w_kernel<<<(D1 * 16) / 256, 256, 0, stream>>>(W1, P1, D1);
  softmax_w_kernel<<<(D2 * 16) / 256, 256, 0, stream>>>(W2, P2, D2);
  softmax_w_kernel<<<(D3 * 16) / 256, 256, 0, stream>>>(W3, P3, D3);
  softmax_w_kernel<<<(D4 * 16) / 256, 256, 0, stream>>>(W4, P4, D4);

  // four cross+logsumexp layers as WMMA GEMMs
  spn_layer_kernel<<<dim3(NSAMP / 16, D1 / 8), 256, 0, stream>>>(x,  P1, h1, D1);
  spn_layer_kernel<<<dim3(NSAMP / 16, D2 / 8), 256, 0, stream>>>(h1, P2, h2, D2);
  spn_layer_kernel<<<dim3(NSAMP / 16, D3 / 8), 256, 0, stream>>>(h2, P3, h3, D3);
  spn_layer_kernel<<<dim3(NSAMP / 16, D4 / 8), 256, 0, stream>>>(h3, P4, h4, D4);

  // root: feature-sum + final logsumexp
  root_kernel<<<NSAMP / 8, 256, 0, stream>>>(h4, Wr, out);
}
